// GCN_GAT_73770358276815
// MI455X (gfx1250) — matrix operand order repaired and checked by reading, not compiled, not run on hardware
//
#include <hip/hip_runtime.h>
#include <hip/hip_bf16.h>
#include <cstdint>

// ---------------- problem constants (match reference) ----------------
#define NN        20000      // nodes
#define NE        320000     // edges
#define IN_CH     256
#define HID       64
#define HEADS     8
#define OUT_CH    64
#define DIM1      (HEADS*HID)   // 512
#define NEG_SLOPE 0.2f

typedef __attribute__((ext_vector_type(16))) _Float16 v16h;
typedef __attribute__((ext_vector_type(8)))  _Float16 v8h;
typedef __attribute__((ext_vector_type(8)))  float    v8f;

// ---------------- small utility kernels ----------------
__global__ void k_fill_f32(float* __restrict__ p, float v, int n) {
  int i = blockIdx.x * blockDim.x + threadIdx.x;
  if (i < n) p[i] = v;
}

__global__ void k_cast_f16(const float* __restrict__ in, _Float16* __restrict__ out, int n) {
  int i = blockIdx.x * blockDim.x + threadIdx.x;
  if (i < n) out[i] = (_Float16)in[i];
}

// W: [K, Ntot] row-major (f32)  ->  WT: [Ntot, K] row-major (f16)
__global__ void k_cast_transpose_f16(const float* __restrict__ W, _Float16* __restrict__ WT,
                                     int K, int Ntot) {
  int i = blockIdx.x * blockDim.x + threadIdx.x;
  if (i >= K * Ntot) return;
  int k = i / Ntot, n = i - k * Ntot;
  WT[(size_t)n * K + k] = (_Float16)W[i];
}

__global__ void k_relu_bias_cast(const float* __restrict__ in, const float* __restrict__ bias,
                                 _Float16* __restrict__ out16, int total, int dim) {
  int i = blockIdx.x * blockDim.x + threadIdx.x;
  if (i >= total) return;
  float v = in[i] + bias[i % dim];
  v = v > 0.f ? v : 0.f;
  out16[i] = (_Float16)v;
}

__global__ void k_bias_add(float* __restrict__ out, const float* __restrict__ bias,
                           int total, int dim) {
  int i = blockIdx.x * blockDim.x + threadIdx.x;
  if (i >= total) return;
  out[i] += bias[i % dim];
}

// ---------------- fused dual-weight WMMA GEMM ----------------
// Cl = A @ Bl, Cr = A @ Br.  A: [M,K] f16 row-major.  BTl/BTr: [Ntot,K] f16
// (weights pre-transposed so each lane's B fragment is one contiguous 32B run).
// One wave32 computes a 32x32 block of BOTH outputs: 2x2 tile blocking x 2
// weight matrices = 8 v_wmma_f32_16x16x32_f16 per K-step fed by only 6
// fragment loads (A fragments reused 4x).
__global__ void __launch_bounds__(256)
k_wmma_gemm_dual(const _Float16* __restrict__ A,
                 const _Float16* __restrict__ BTl, const _Float16* __restrict__ BTr,
                 float* __restrict__ Cl, float* __restrict__ Cr,
                 int M, int Ntot, int K) {
  const int lane   = threadIdx.x & 31;
  const int waveId = blockIdx.x * (blockDim.x >> 5) + (threadIdx.x >> 5);
  const int tilesN = Ntot >> 5;                    // 32-wide super-tiles
  const int tm = waveId / tilesN;
  const int tn = waveId - tm * tilesN;
  if (tm >= (M >> 5)) return;   // wave-uniform guard; EXEC all-ones at WMMAs

  const int hi = lane >> 4;                        // half-wave select
  const int r  = (tm << 5) + (lane & 15);          // A row (sub-tile 0)
  const int c  = (tn << 5) + (lane & 15);          // B col (sub-tile 0)
  const size_t aOff0 = (size_t)r * K;
  const size_t aOff1 = aOff0 + (size_t)16 * K;
  const size_t bOff0 = (size_t)c * K;
  const size_t bOff1 = bOff0 + (size_t)16 * K;

  v8f l00 = {}, l01 = {}, l10 = {}, l11 = {};
  v8f r00 = {}, r01 = {}, r10 = {}, r11 = {};

  for (int k0 = 0; k0 < K; k0 += 32) {
    // A 16x32 fragments (contiguous k order, same as B): lanes 0-15 K=[k0..k0+15],
    // lanes 16-31 K=[k0+16..k0+31], row = lane%16.
    const int ka = k0 + (hi << 4);
    union { v16h v; v8h h[2]; } a0, a1;
    a0.h[0] = *(const v8h*)(A + aOff0 + ka);
    a0.h[1] = *(const v8h*)(A + aOff0 + ka + 8);
    a1.h[0] = *(const v8h*)(A + aOff1 + ka);
    a1.h[1] = *(const v8h*)(A + aOff1 + ka + 8);
    __builtin_prefetch(A + aOff0 + ka + 32, 0, 3);
    __builtin_prefetch(A + aOff1 + ka + 32, 0, 3);
    // B 32x16 fragments: lane holds col = lane%16, K = k0 + hi*16 .. +15
    const int kb = k0 + (hi << 4);
    v16h bl0 = *(const v16h*)(BTl + bOff0 + kb);
    v16h bl1 = *(const v16h*)(BTl + bOff1 + kb);
    v16h br0 = *(const v16h*)(BTr + bOff0 + kb);
    v16h br1 = *(const v16h*)(BTr + bOff1 + kb);

    l00 = __builtin_amdgcn_wmma_f32_16x16x32_f16(false, a0.v, false, bl0, (short)0, l00, false, false);
    l01 = __builtin_amdgcn_wmma_f32_16x16x32_f16(false, a0.v, false, bl1, (short)0, l01, false, false);
    l10 = __builtin_amdgcn_wmma_f32_16x16x32_f16(false, a1.v, false, bl0, (short)0, l10, false, false);
    l11 = __builtin_amdgcn_wmma_f32_16x16x32_f16(false, a1.v, false, bl1, (short)0, l11, false, false);
    r00 = __builtin_amdgcn_wmma_f32_16x16x32_f16(false, a0.v, false, br0, (short)0, r00, false, false);
    r01 = __builtin_amdgcn_wmma_f32_16x16x32_f16(false, a0.v, false, br1, (short)0, r01, false, false);
    r10 = __builtin_amdgcn_wmma_f32_16x16x32_f16(false, a1.v, false, br0, (short)0, r10, false, false);
    r11 = __builtin_amdgcn_wmma_f32_16x16x32_f16(false, a1.v, false, br1, (short)0, r11, false, false);
  }

  // C/D layout: VGPR v -> M = mBase + v + (hi ? 8 : 0), N = nBase + lane%16
  const int m0 = (tm << 5) + (hi << 3);
  float* Clp = Cl + (size_t)m0 * Ntot + c;
  float* Crp = Cr + (size_t)m0 * Ntot + c;
  const size_t rstep = (size_t)Ntot;
#pragma unroll
  for (int v = 0; v < 8; ++v) {
    Clp[(size_t)v * rstep]               = l00[v];
    Clp[(size_t)v * rstep + 16]          = l01[v];
    Clp[((size_t)v + 16) * rstep]        = l10[v];
    Clp[((size_t)v + 16) * rstep + 16]   = l11[v];
    Crp[(size_t)v * rstep]               = r00[v];
    Crp[(size_t)v * rstep + 16]          = r01[v];
    Crp[((size_t)v + 16) * rstep]        = r10[v];
    Crp[((size_t)v + 16) * rstep + 16]   = r11[v];
  }
}

// ---------------- edge kernels ----------------
__device__ __forceinline__ void atomicMaxF(float* addr, float val) {
  if (val >= 0.f) atomicMax((int*)addr, __float_as_int(val));
  else            atomicMin((unsigned int*)addr, __float_as_uint(val));
}

// one wave32 per (edge, head): alpha[e,h] = att_h . leaky_relu(xl[src]+xr[dst]); amax atomic-max
__global__ void k_edge_scores(const float* __restrict__ xl, const float* __restrict__ xr,
                              const float* __restrict__ att,
                              const int* __restrict__ src, const int* __restrict__ dst,
                              float* __restrict__ alpha, float* __restrict__ amax,
                              int E, int heads, int C, int dim) {
  int gid  = blockIdx.x * blockDim.x + threadIdx.x;
  int wave = gid >> 5, lane = gid & 31;
  if (wave >= E * heads) return;
  int e = wave / heads, h = wave - e * heads;
  int s = src[e], d = dst[e];
  const float* pl = xl + (size_t)s * dim + h * C;
  const float* pr = xr + (size_t)d * dim + h * C;
  const float* pa = att + h * C;
  float acc = 0.f;
  for (int c = lane; c < C; c += 32) {
    float v = pl[c] + pr[c];
    v = v > 0.f ? v : v * NEG_SLOPE;
    acc += v * pa[c];
  }
#pragma unroll
  for (int off = 16; off > 0; off >>= 1) acc += __shfl_xor(acc, off, 32);
  if (lane == 0) {
    alpha[wave] = acc;
    atomicMaxF(&amax[(size_t)d * heads + h], acc);
  }
}

// one thread per (edge, head): alpha <- exp(alpha - amax[dst]); denom atomic-add
__global__ void k_edge_exp(const int* __restrict__ dst, float* __restrict__ alpha,
                           const float* __restrict__ amax, float* __restrict__ den,
                           int E, int heads) {
  int i = blockIdx.x * blockDim.x + threadIdx.x;
  if (i >= E * heads) return;
  int e = i / heads, h = i - e * heads;
  int d = dst[e];
  float ex = __expf(alpha[i] - amax[(size_t)d * heads + h]);
  alpha[i] = ex;
  atomicAdd(&den[(size_t)d * heads + h], ex);
}

// one wave32 per (edge, head): agg[dst] += xl[src] * (ex / denom)
__global__ void k_edge_aggregate(const float* __restrict__ xl, const int* __restrict__ src,
                                 const int* __restrict__ dst, const float* __restrict__ alpha,
                                 const float* __restrict__ den, float* __restrict__ agg,
                                 int E, int heads, int C, int dim) {
  int gid  = blockIdx.x * blockDim.x + threadIdx.x;
  int wave = gid >> 5, lane = gid & 31;
  if (wave >= E * heads) return;
  int e = wave / heads, h = wave - e * heads;
  int s = src[e], d = dst[e];
  float w = alpha[wave] / (den[(size_t)d * heads + h] + 1e-16f);
  const float* pl = xl + (size_t)s * dim + h * C;
  float* po = agg + (size_t)d * dim + h * C;
  for (int c = lane; c < C; c += 32) atomicAdd(&po[c], pl[c] * w);
}

// ---------------- host driver ----------------
static inline int ceil_div(long long a, long long b) { return (int)((a + b - 1) / b); }

extern "C" void kernel_launch(void* const* d_in, const int* in_sizes, int n_in,
                              void* d_out, int out_size, void* d_ws, size_t ws_size,
                              hipStream_t stream) {
  (void)in_sizes; (void)n_in; (void)out_size; (void)ws_size;
  const float* feat = (const float*)d_in[0];
  const int*   ei   = (const int*)d_in[1];
  const float* Wl1  = (const float*)d_in[2];
  const float* Wr1  = (const float*)d_in[3];
  const float* att1 = (const float*)d_in[4];
  const float* b1   = (const float*)d_in[5];
  const float* Wl2  = (const float*)d_in[6];
  const float* Wr2  = (const float*)d_in[7];
  const float* att2 = (const float*)d_in[8];
  const float* b2   = (const float*)d_in[9];
  const int* src = ei;
  const int* dst = ei + NE;
  float* out = (float*)d_out;

  // --- workspace bump allocation (256B aligned), with later aliasing ---
  char* p = (char*)d_ws;
  auto alloc = [&p](size_t bytes) -> char* {
    uintptr_t v = ((uintptr_t)p + 255) & ~(uintptr_t)255;
    p = (char*)v + bytes;
    return (char*)v;
  };
  _Float16* xf16  = (_Float16*)alloc((size_t)NN * IN_CH * 2);     // 10.24 MB
  _Float16* w1lT  = (_Float16*)alloc((size_t)DIM1 * IN_CH * 2);
  _Float16* w1rT  = (_Float16*)alloc((size_t)DIM1 * IN_CH * 2);
  float*    xl1   = (float*)   alloc((size_t)NN * DIM1 * 4);      // 40.96 MB
  float*    xr1   = (float*)   alloc((size_t)NN * DIM1 * 4);      // 40.96 MB
  float*    alpha1= (float*)   alloc((size_t)NE * HEADS * 4);     // 10.24 MB
  float*    amax1 = (float*)   alloc((size_t)NN * HEADS * 4);
  float*    den1  = (float*)   alloc((size_t)NN * HEADS * 4);
  _Float16* w2lT  = (_Float16*)alloc((size_t)OUT_CH * DIM1 * 2);
  _Float16* w2rT  = (_Float16*)alloc((size_t)OUT_CH * DIM1 * 2);
  // aliases (lifetimes disjoint):
  float*    agg1  = xr1;                                  // after L1 scores, xr1 is dead
  _Float16* h16   = (_Float16*)xl1;                       // after L1 aggregate, xl1 is dead
  float*    xl2   = (float*)xf16;                         // after L1 GEMMs, xf16 is dead
  float*    xr2   = (float*)((char*)xf16 + (size_t)NN * OUT_CH * 4);
  float*    alpha2= alpha1;                               // after L1 aggregate, alpha1 dead
  float*    amax2 = alpha1 + NE;
  float*    den2  = amax2 + NN;

  const int T = 256;
  // ---- casts / transposes ----
  k_cast_f16<<<ceil_div((long long)NN*IN_CH, T), T, 0, stream>>>(feat, xf16, NN*IN_CH);
  k_cast_transpose_f16<<<ceil_div((long long)IN_CH*DIM1, T), T, 0, stream>>>(Wl1, w1lT, IN_CH, DIM1);
  k_cast_transpose_f16<<<ceil_div((long long)IN_CH*DIM1, T), T, 0, stream>>>(Wr1, w1rT, IN_CH, DIM1);
  k_cast_transpose_f16<<<ceil_div((long long)DIM1*OUT_CH, T), T, 0, stream>>>(Wl2, w2lT, DIM1, OUT_CH);
  k_cast_transpose_f16<<<ceil_div((long long)DIM1*OUT_CH, T), T, 0, stream>>>(Wr2, w2rT, DIM1, OUT_CH);
  // ---- layer 1 projections: xl1 = x@Wl1, xr1 = x@Wr1, fused (WMMA) ----
  {
    int waves  = (NN / 32) * (DIM1 / 32);          // 625 * 16 = 10000 waves
    int blocks = ceil_div((long long)waves * 32, T);
    k_wmma_gemm_dual<<<blocks, T, 0, stream>>>(xf16, w1lT, w1rT, xl1, xr1, NN, DIM1, IN_CH);
  }
  // ---- layer 1 attention ----
  k_fill_f32<<<ceil_div((long long)NN*HEADS, T), T, 0, stream>>>(amax1, -INFINITY, NN*HEADS);
  k_fill_f32<<<ceil_div((long long)NN*HEADS, T), T, 0, stream>>>(den1, 0.f, NN*HEADS);
  {
    long long waves = (long long)NE * HEADS;
    k_edge_scores<<<ceil_div(waves*32, T), T, 0, stream>>>(xl1, xr1, att1, src, dst,
                                                          alpha1, amax1, NE, HEADS, HID, DIM1);
    k_fill_f32<<<ceil_div((long long)NN*DIM1, T), T, 0, stream>>>(agg1, 0.f, NN*DIM1);
    k_edge_exp<<<ceil_div(waves, T), T, 0, stream>>>(dst, alpha1, amax1, den1, NE, HEADS);
    k_edge_aggregate<<<ceil_div(waves*32, T), T, 0, stream>>>(xl1, src, dst, alpha1, den1,
                                                             agg1, NE, HEADS, HID, DIM1);
  }
  // ---- h = relu(agg1 + b1), cast f16 ----
  k_relu_bias_cast<<<ceil_div((long long)NN*DIM1, T), T, 0, stream>>>(agg1, b1, h16, NN*DIM1, DIM1);
  // ---- layer 2 projections: xl2 = h@Wl2, xr2 = h@Wr2, fused (WMMA) ----
  {
    int waves  = (NN / 32) * (OUT_CH / 32);        // 625 * 2 = 1250 waves
    int blocks = ceil_div((long long)waves * 32, T);
    k_wmma_gemm_dual<<<blocks, T, 0, stream>>>(h16, w2lT, w2rT, xl2, xr2, NN, OUT_CH, DIM1);
  }
  // ---- layer 2 attention ----
  k_fill_f32<<<ceil_div((long long)NN, T), T, 0, stream>>>(amax2, -INFINITY, NN);
  k_fill_f32<<<ceil_div((long long)NN, T), T, 0, stream>>>(den2, 0.f, NN);
  k_fill_f32<<<ceil_div((long long)NN*OUT_CH, T), T, 0, stream>>>(out, 0.f, NN*OUT_CH);
  {
    long long waves = NE;
    k_edge_scores<<<ceil_div(waves*32, T), T, 0, stream>>>(xl2, xr2, att2, src, dst,
                                                          alpha2, amax2, NE, 1, OUT_CH, OUT_CH);
    k_edge_exp<<<ceil_div(waves, T), T, 0, stream>>>(dst, alpha2, amax2, den2, NE, 1);
    k_edge_aggregate<<<ceil_div(waves*32, T), T, 0, stream>>>(xl2, src, dst, alpha2, den2,
                                                             out, NE, 1, OUT_CH, OUT_CH);
  }
  k_bias_add<<<ceil_div((long long)NN*OUT_CH, T), T, 0, stream>>>(out, b2, NN*OUT_CH, OUT_CH);
}
